// MultiHeadAttention_10333691314605
// MI455X (gfx1250) — matrix operand order repaired and checked by reading, not compiled, not run on hardware
//
#include <hip/hip_runtime.h>
#include <hip/hip_bf16.h>
#include <math.h>

#define DEV __device__ __forceinline__

typedef __attribute__((ext_vector_type(16))) _Float16 v16h;
typedef __attribute__((ext_vector_type(8)))  _Float16 v8h;
typedef __attribute__((ext_vector_type(4)))  _Float16 v4h;
typedef __attribute__((ext_vector_type(8)))  float    v8f;

union Frag16 { v16h v; v8h h[2]; };

constexpr int Bsz  = 2;
constexpr int Sl   = 2048;
constexpr int Dm   = 1024;
constexpr int Hn   = 16;
constexpr int Hd   = 64;
constexpr int Mtot = Bsz * Sl;   // 4096 tokens
constexpr int BH   = Bsz * Hn;   // 32

// ---------------------------------------------------------------------------
// WMMA helpers (CDNA5 wave32, 16x16x32 f16 -> f32)
// ---------------------------------------------------------------------------
DEV v8f wmma_f16(v16h a, v16h b, v8f c) {
  // (neg_a, A, neg_b, B, c_mod, C, reuse_a, reuse_b)
  return __builtin_amdgcn_wmma_f32_16x16x32_f16(false, a, false, b, (short)0, c,
                                                false, false);
}

// A fragment 16x32 f16: lane row m = lane&15; lanes 0-15: halves0-7 -> K=0..7,
// halves8-15 -> K=16..23; lanes 16-31 offset K by +8.  rowbase points at
// &A[m][k0]; both 8-half chunks are contiguous -> 2x global_load_b128 / lane.
DEV v16h load_a(const _Float16* rowbase, int laneHi) {
  Frag16 f;
  f.h[0] = *(const v8h*)(rowbase + 8 * laneHi);
  f.h[1] = *(const v8h*)(rowbase + 16 + 8 * laneHi);
  return f.v;
}

// B fragment 32x16 f16: lane col n = lane&15; lane holds K chunks at kbase and
// kbase+16 (kbase = &Bcol[k0 + 8*laneHi]) -> 2x global_load_b128 / lane.
DEV v16h load_b(const _Float16* kbase) {
  Frag16 f;
  f.h[0] = *(const v8h*)(kbase);
  f.h[1] = *(const v8h*)(kbase + 16);
  return f.v;
}

DEV float red16_max(float v) {
  v = fmaxf(v, __shfl_xor(v, 1));
  v = fmaxf(v, __shfl_xor(v, 2));
  v = fmaxf(v, __shfl_xor(v, 4));
  v = fmaxf(v, __shfl_xor(v, 8));
  return v;
}
DEV float red16_sum(float v) {
  v += __shfl_xor(v, 1);
  v += __shfl_xor(v, 2);
  v += __shfl_xor(v, 4);
  v += __shfl_xor(v, 8);
  return v;
}

// ---------------------------------------------------------------------------
// f32 -> f16 conversion (vectorized: 4 elems / thread)
// ---------------------------------------------------------------------------
__global__ __launch_bounds__(256) void cvt_f32_f16(const float* __restrict__ in,
                                                   _Float16* __restrict__ out,
                                                   int n4) {
  int i = blockIdx.x * blockDim.x + threadIdx.x;
  if (i >= n4) return;
  float4 v = ((const float4*)in)[i];
  v4h o;
  o[0] = (_Float16)v.x; o[1] = (_Float16)v.y;
  o[2] = (_Float16)v.z; o[3] = (_Float16)v.w;
  ((v4h*)out)[i] = o;
}

// ---------------------------------------------------------------------------
// GEMM  y = A[M,K] @ W[N,K]^T   (both row-major; per-lane b128 fragments)
// wave tile: 32(M) x 64(N), 4 waves/block -> block tile 32 x 256.
// MODE 0: y -> f16 [b,h,s,hd]   (Q and K)
// MODE 2: y -> f16 [b,h,hd,s]   (V transposed for the PV B-fragments)
// MODE 3: y -> f32 [m,n] + bias (output projection)
// Compile-time MODE => straight-line epilogue, no per-element branching.
// ---------------------------------------------------------------------------
template <int MODE>
__global__ __launch_bounds__(128) void gemm_xwt(
    const _Float16* __restrict__ A, const _Float16* __restrict__ W,
    _Float16* __restrict__ Oh, float* __restrict__ Of,
    const float* __restrict__ bias) {
  const int K      = Dm;
  const int lane   = threadIdx.x & 31;
  const int wave   = threadIdx.x >> 5;
  const int laneHi = lane >> 4;
  const int laneLo = lane & 15;

  const int m0 = blockIdx.x * 32;
  const int n0 = blockIdx.y * 256 + wave * 64;

  v8f acc[2][4] = {};
  for (int k0 = 0; k0 < K; k0 += 32) {
    v16h af[2], bf[4];
#pragma unroll
    for (int i = 0; i < 2; ++i)
      af[i] = load_a(A + (size_t)(m0 + 16 * i + laneLo) * K + k0, laneHi);
#pragma unroll
    for (int j = 0; j < 4; ++j)
      bf[j] = load_b(W + (size_t)(n0 + 16 * j + laneLo) * K + k0 + 8 * laneHi);
#pragma unroll
    for (int i = 0; i < 2; ++i)
#pragma unroll
      for (int j = 0; j < 4; ++j)
        acc[i][j] = wmma_f16(af[i], bf[j], acc[i][j]);
  }

  // Epilogue: C layout -> lane holds rows m = r + 8*laneHi, col n = laneLo.
#pragma unroll
  for (int i = 0; i < 2; ++i) {
#pragma unroll
    for (int j = 0; j < 4; ++j) {
#pragma unroll
      for (int r = 0; r < 8; ++r) {
        int m = m0 + 16 * i + 8 * laneHi + r;
        int n = n0 + 16 * j + laneLo;
        float val = acc[i][j][r];
        if constexpr (MODE == 3) {
          Of[(size_t)m * Dm + n] = val + bias[n];
        } else {
          int bb = m >> 11, s = m & (Sl - 1);   // S = 2048
          int h = n >> 6, hd = n & (Hd - 1);
          if constexpr (MODE == 2)
            Oh[(((size_t)(bb * Hn + h)) * Hd + hd) * Sl + s] = (_Float16)val;
          else
            Oh[(((size_t)(bb * Hn + h)) * Sl + s) * Hd + hd] = (_Float16)val;
        }
      }
    }
  }
}

// ---------------------------------------------------------------------------
// Causal flash attention. One wave owns one 16-query tile of one (b,h).
// 64 keys / iteration: 4 score tiles (8 chained QK^T WMMAs), one online-
// softmax pass, P transposed C->A layout through a private LDS slice
// (ds_store_b16 -> ds_load_b128), then 8 PV WMMAs.
// NOTE: no __syncthreads — waves have different causal trip counts.
// ---------------------------------------------------------------------------
__global__ __launch_bounds__(128) void flash_attn(
    const _Float16* __restrict__ Q,    // [BH, S, Hd]
    const _Float16* __restrict__ Kt,   // [BH, S, Hd]
    const _Float16* __restrict__ Vt,   // [BH, Hd, S]
    _Float16* __restrict__ ctx) {      // [B, S, D]
  constexpr int LSTRIDE = 72;                  // halves; 144B row, 16B-aligned
  __shared__ _Float16 lds[4][16 * LSTRIDE];

  const int lane   = threadIdx.x & 31;
  const int wave   = threadIdx.x >> 5;
  const int laneHi = lane >> 4;
  const int laneLo = lane & 15;

  const int gw = blockIdx.x * 4 + wave;    // 0 .. BH*S/16-1
  const int qt = gw & (Sl / 16 - 1);       // 128 q-tiles per head
  const int bh = gw >> 7;
  const int q0 = qt * 16;

  const _Float16* Qb = Q  + (size_t)bh * Sl * Hd;
  const _Float16* Kb = Kt + (size_t)bh * Sl * Hd;
  const _Float16* Vb = Vt + (size_t)bh * Hd * Sl;
  _Float16* Lw = lds[wave];

  // Q fragments (hd split 0..31 / 32..63), 1/sqrt(64) folded in.
  v16h qa[2];
#pragma unroll
  for (int c = 0; c < 2; ++c) {
    Frag16 f;
    const _Float16* base = Qb + (size_t)(q0 + laneLo) * Hd + c * 32;
    f.h[0] = *(const v8h*)(base + 8 * laneHi);
    f.h[1] = *(const v8h*)(base + 16 + 8 * laneHi);
#pragma unroll
    for (int e = 0; e < 16; ++e)
      f.v[e] = (_Float16)((float)f.v[e] * 0.125f);
    qa[c] = f.v;
  }

  v8f acc[4] = {};
  float mrun[8], lrun[8];
#pragma unroll
  for (int r = 0; r < 8; ++r) { mrun[r] = -INFINITY; lrun[r] = 0.0f; }

  // j0 steps by 64; largest j0 is <= q0 (and <= 1984), so key loads stay
  // in-bounds (keys <= 2047) and every row keeps >= 1 unmasked key.
  const int jend = q0 + 16;                // keys < jend (causal)
  for (int j0 = 0; j0 < jend; j0 += 64) {
    // ---- scores = (Q/8) @ K^T for 64 keys -------------------------------
    v8f s[4] = {};
#pragma unroll
    for (int t = 0; t < 4; ++t)
#pragma unroll
      for (int c = 0; c < 2; ++c) {
        const _Float16* kb =
            Kb + (size_t)(j0 + t * 16 + laneLo) * Hd + c * 32 + 8 * laneHi;
        s[t] = wmma_f16(qa[c], load_b(kb), s[t]);
      }

    // ---- causal mask (only diagonal-crossing passes) --------------------
    if (j0 + 63 > q0) {
#pragma unroll
      for (int t = 0; t < 4; ++t)
#pragma unroll
        for (int r = 0; r < 8; ++r) {
          int key   = j0 + t * 16 + laneLo;
          int query = q0 + 8 * laneHi + r;
          if (key > query) s[t][r] = -INFINITY;
        }
    }

    // ---- online softmax update + stage P (f16) into LDS -----------------
    float alpha[8];
#pragma unroll
    for (int r = 0; r < 8; ++r) {
      float mx = fmaxf(fmaxf(s[0][r], s[1][r]), fmaxf(s[2][r], s[3][r]));
      mx = red16_max(mx);
      float mnew = fmaxf(mrun[r], mx);
      alpha[r] = __expf(mrun[r] - mnew);
      int row = 8 * laneHi + r;
      float ps = 0.0f;
#pragma unroll
      for (int t = 0; t < 4; ++t) {
        float p = __expf(s[t][r] - mnew);
        Lw[row * LSTRIDE + t * 16 + laneLo] = (_Float16)p;
        ps += p;
      }
      lrun[r] = lrun[r] * alpha[r] + red16_sum(ps);
      mrun[r] = mnew;
    }
#pragma unroll
    for (int j = 0; j < 4; ++j)
#pragma unroll
      for (int r = 0; r < 8; ++r) acc[j][r] *= alpha[r];

    // ---- P: C-layout -> A-layout via LDS (same-wave ds RAW) -------------
    v16h pa[2];
    const _Float16* prow = Lw + laneLo * LSTRIDE;
#pragma unroll
    for (int c = 0; c < 2; ++c) {
      Frag16 pf;
      pf.h[0] = *(const v8h*)(prow + c * 32 + 8 * laneHi);
      pf.h[1] = *(const v8h*)(prow + c * 32 + 16 + 8 * laneHi);
      pa[c] = pf.v;
    }

    // ---- ctx += P @ V  (V^T layout makes B-columns contiguous) ----------
#pragma unroll
    for (int j = 0; j < 4; ++j)
#pragma unroll
      for (int c = 0; c < 2; ++c) {
        const _Float16* vb =
            Vb + (size_t)(j * 16 + laneLo) * Sl + j0 + c * 32 + 8 * laneHi;
        acc[j] = wmma_f16(pa[c], load_b(vb), acc[j]);
      }
  }

  // ---- normalize (v_rcp) + write ctx f16 [b, s, h*64+hd] ----------------
  float rl[8];
#pragma unroll
  for (int r = 0; r < 8; ++r) rl[r] = __builtin_amdgcn_rcpf(lrun[r]);

  const int bb = bh >> 4;        // / Hn
  const int h  = bh & (Hn - 1);
#pragma unroll
  for (int j = 0; j < 4; ++j)
#pragma unroll
    for (int r = 0; r < 8; ++r) {
      int tok = q0 + 8 * laneHi + r;
      int col = h * Hd + j * 16 + laneLo;
      float o = acc[j][r] * rl[r];
      ctx[((size_t)(bb * Sl + tok)) * Dm + col] = (_Float16)o;
    }
}

// ---------------------------------------------------------------------------
// Host-side launch
// ---------------------------------------------------------------------------
extern "C" void kernel_launch(void* const* d_in, const int* in_sizes, int n_in,
                              void* d_out, int out_size, void* d_ws,
                              size_t ws_size, hipStream_t stream) {
  (void)in_sizes; (void)n_in; (void)out_size; (void)ws_size;
  const float* x  = (const float*)d_in[0];
  const float* wq = (const float*)d_in[1];
  const float* wk = (const float*)d_in[2];
  const float* wv = (const float*)d_in[3];
  const float* wo = (const float*)d_in[4];
  const float* bo = (const float*)d_in[5];
  float* out = (float*)d_out;

  const size_t nMD = (size_t)Mtot * Dm;   // 4,194,304
  const size_t nDD = (size_t)Dm * Dm;     // 1,048,576

  _Float16* ws   = (_Float16*)d_ws;
  _Float16* xh   = ws;
  _Float16* wqh  = xh  + nMD;
  _Float16* wkh  = wqh + nDD;
  _Float16* wvh  = wkh + nDD;
  _Float16* woh  = wvh + nDD;
  _Float16* Qh   = woh + nDD;
  _Float16* Kh   = Qh  + nMD;
  _Float16* Vth  = Kh  + nMD;
  _Float16* ctxh = Vth + nMD;   // total 48 MB of workspace

  // f32 -> f16
  cvt_f32_f16<<<(int)(nMD / 4 / 256), 256, 0, stream>>>(x,  xh,  (int)(nMD / 4));
  cvt_f32_f16<<<(int)(nDD / 4 / 256), 256, 0, stream>>>(wq, wqh, (int)(nDD / 4));
  cvt_f32_f16<<<(int)(nDD / 4 / 256), 256, 0, stream>>>(wk, wkh, (int)(nDD / 4));
  cvt_f32_f16<<<(int)(nDD / 4 / 256), 256, 0, stream>>>(wv, wvh, (int)(nDD / 4));
  cvt_f32_f16<<<(int)(nDD / 4 / 256), 256, 0, stream>>>(wo, woh, (int)(nDD / 4));

  // QKV projections
  dim3 ggrid(Mtot / 32, Dm / 256);
  gemm_xwt<0><<<ggrid, 128, 0, stream>>>(xh, wqh, Qh,  nullptr, nullptr);
  gemm_xwt<0><<<ggrid, 128, 0, stream>>>(xh, wkh, Kh,  nullptr, nullptr);
  gemm_xwt<2><<<ggrid, 128, 0, stream>>>(xh, wvh, Vth, nullptr, nullptr);

  // causal flash attention: BH * S/16 = 4096 waves, 4 waves/block
  flash_attn<<<dim3(BH * (Sl / 16) / 4), 128, 0, stream>>>(Qh, Kh, Vth, ctxh);

  // output projection + bias (f32 out)
  gemm_xwt<3><<<ggrid, 128, 0, stream>>>(ctxh, woh, nullptr, out, bo);
}
